// GCN_72164040507601
// MI455X (gfx1250) — compile-verified, matrix operand order checked by reading, not run on hardware
//
#include <hip/hip_runtime.h>
#include <hip/hip_bf16.h>
#include <stdint.h>

// ---------------------------------------------------------------------------
// CDNA5 (gfx1250) GCN forward. wave32, WMMA bf16 16x16x32 for dense GEMMs,
// vectorized b128 traffic + f32 atomics for the bandwidth-bound edge scatter.
// ---------------------------------------------------------------------------

typedef __attribute__((ext_vector_type(16))) __bf16 v16bf;
typedef __attribute__((ext_vector_type(8)))  float  v8f;

struct __align__(16) U4  { unsigned int a, b, c, d; };
struct __align__(8)  US4 { unsigned short x, y, z, w; };
union FragBF { v16bf v; U4 q[2]; unsigned short u[16]; };

__device__ __forceinline__ unsigned short f2bf_bits(float x) {
    unsigned int u = __float_as_uint(x);
    unsigned int r = u + 0x7FFFu + ((u >> 16) & 1u);   // RNE to bf16
    return (unsigned short)(r >> 16);
}

// ---------------------------------------------------------------------------
// Elementwise / setup kernels (16B-vectorized; all sizes are multiples of 4)
// ---------------------------------------------------------------------------

__global__ void zero_f32_v4(float4* __restrict__ p, int n4) {
    int i = blockIdx.x * blockDim.x + threadIdx.x;
    if (i < n4) p[i] = make_float4(0.f, 0.f, 0.f, 0.f);
}

__global__ void deg_count(const int* __restrict__ dst, float* __restrict__ deg, int E) {
    int e = blockIdx.x * blockDim.x + threadIdx.x;
    if (e < E) atomicAdd(&deg[dst[e]], 1.0f);
}

__global__ void deg_rsqrt(float* __restrict__ deg, int n) {
    int i = blockIdx.x * blockDim.x + threadIdx.x;
    if (i < n) deg[i] = rsqrtf(deg[i] + 1.0f);   // include self loop
}

__global__ void cvt_f32_bf16_v4(const float4* __restrict__ in, US4* __restrict__ out, int n4) {
    int i = blockIdx.x * blockDim.x + threadIdx.x;
    if (i >= n4) return;
    float4 v = in[i];
    US4 o;
    o.x = f2bf_bits(v.x); o.y = f2bf_bits(v.y);
    o.z = f2bf_bits(v.z); o.w = f2bf_bits(v.w);
    out[i] = o;
}

// Pack W[K][64] (row-major f32) into per-lane-contiguous bf16 B-fragments:
// layout [kt][nt][lane][i] with i the v16bf element index per ISA 7.12.2.
__global__ void pack_W(const float* __restrict__ W, unsigned short* __restrict__ Wp, int K) {
    int idx = blockIdx.x * blockDim.x + threadIdx.x;
    int total = (K >> 5) * 4 * 32 * 16;
    if (idx >= total) return;
    int i    = idx & 15;
    int lane = (idx >> 4) & 31;
    int nt   = (idx >> 9) & 3;
    int kt   = idx >> 11;
    int h    = lane >> 4;
    int col  = nt * 16 + (lane & 15);
    int k    = kt * 32 + h * 8 + (i & 7) + ((i >> 3) << 4);
    Wp[idx] = f2bf_bits(W[k * 64 + col]);
}

// ---------------------------------------------------------------------------
// WMMA GEMM: H[M][64] = A_bf16[M][K] @ Wpack  (f32 accumulate)
// One wave per 16x64 output tile; 8 waves (256 thr) per block -> 128 rows.
// ---------------------------------------------------------------------------

__global__ __launch_bounds__(256) void gemm_wmma_bf16(
    const unsigned short* __restrict__ Abf,   // [M][K] bf16 bits, row-major
    const unsigned short* __restrict__ Wpack, // packed fragments
    float* __restrict__ Hout,                 // [M][64] f32
    int M, int K)
{
    const int wave = threadIdx.x >> 5;
    const int lane = threadIdx.x & 31;
    const int row_base = (blockIdx.x * 8 + wave) * 16;
    if (row_base >= M) return;                // wave-uniform: EXEC stays all-ones

    const int h = lane >> 4;                  // K-half select
    const int r = lane & 15;                  // row within tile / col within n-tile

    v8f acc[4] = {};                          // 4 n-tiles of 16 cols

    const int ksteps = K >> 5;
    for (int kt = 0; kt < ksteps; ++kt) {
        const int kk = kt * 32;
        FragBF af;
        const unsigned short* ap = Abf + (size_t)(row_base + r) * K + kk + h * 8;
        af.q[0] = *reinterpret_cast<const U4*>(ap);       // K = kk+h*8 .. +7
        af.q[1] = *reinterpret_cast<const U4*>(ap + 16);  // K = kk+16+h*8 .. +7
#pragma unroll
        for (int nt = 0; nt < 4; ++nt) {
            FragBF bfm;
            const unsigned short* bp = Wpack + (((size_t)kt * 4 + nt) * 32 + lane) * 16;
            bfm.q[0] = *reinterpret_cast<const U4*>(bp);
            bfm.q[1] = *reinterpret_cast<const U4*>(bp + 8);
            acc[nt] = __builtin_amdgcn_wmma_f32_16x16x32_bf16(
                false, af.v, false, bfm.v, (short)0, acc[nt], false, false);
        }
    }

    // C/D layout: elem e of acc[nt] -> (row_base + e + 8*h, nt*16 + r)
#pragma unroll
    for (int nt = 0; nt < 4; ++nt) {
#pragma unroll
        for (int e = 0; e < 8; ++e) {
            Hout[(size_t)(row_base + e + 8 * h) * 64 + nt * 16 + r] = acc[nt][e];
        }
    }
}

// ---------------------------------------------------------------------------
// Edge scatter: agg[dst] += h[src] * dis[src]*dis[dst]   (bandwidth-bound)
// One thread per (edge, 16B chunk): one b128 gather + 4 f32 atomics.
// 16 consecutive threads cover one edge's 256B feature row.
// ---------------------------------------------------------------------------

__global__ __launch_bounds__(256) void edge_scatter(
    const int* __restrict__ src, const int* __restrict__ dst,
    const float* __restrict__ dis, const float* __restrict__ h,
    float* __restrict__ agg, int E)
{
    int idx = blockIdx.x * blockDim.x + threadIdx.x;   // E*16 = 51.2M < 2^31
    if (idx >= E * 16) return;
    int e = idx >> 4;
    int c = (idx & 15) << 2;                           // feature base 0,4,...,60
    int s = src[e];
    int d = dst[e];
    float norm = dis[s] * dis[d];
    float4 hv = *reinterpret_cast<const float4*>(h + (size_t)s * 64 + c);
    float* ab = agg + (size_t)d * 64 + c;
    atomicAdd(ab + 0, hv.x * norm);
    atomicAdd(ab + 1, hv.y * norm);
    atomicAdd(ab + 2, hv.z * norm);
    atomicAdd(ab + 3, hv.w * norm);
}

// ---------------------------------------------------------------------------
// Fused epilogue: v = relu(agg + h*dis^2 + b), 16B vectorized.
// mode 0: write packed bf16 (input to next WMMA GEMM). mode 1: f32 in-place.
// ---------------------------------------------------------------------------

__global__ void epilogue_relu_v4(
    const float4* __restrict__ h, float4* __restrict__ agg,
    const float* __restrict__ dis, const float* __restrict__ b,
    US4* __restrict__ out_bf, int n_nodes, int mode)
{
    int i = blockIdx.x * blockDim.x + threadIdx.x;     // over n_nodes*16
    if (i >= n_nodes * 16) return;
    int n  = i >> 4;
    int fb = (i & 15) << 2;
    float ds = dis[n];
    float dss = ds * ds;
    float4 hv = h[i];
    float4 av = agg[i];
    float v0 = fmaxf(av.x + hv.x * dss + b[fb + 0], 0.f);
    float v1 = fmaxf(av.y + hv.y * dss + b[fb + 1], 0.f);
    float v2 = fmaxf(av.z + hv.z * dss + b[fb + 2], 0.f);
    float v3 = fmaxf(av.w + hv.w * dss + b[fb + 3], 0.f);
    if (mode == 0) {
        US4 o;
        o.x = f2bf_bits(v0); o.y = f2bf_bits(v1);
        o.z = f2bf_bits(v2); o.w = f2bf_bits(v3);
        out_bf[i] = o;
    } else {
        agg[i] = make_float4(v0, v1, v2, v3);
    }
}

// ---------------------------------------------------------------------------
// Global mean pool: batch is sorted, so do a segmented register accumulation
// per block (64 threads = 64 features, 128 nodes per block) and only emit an
// atomic when the graph id changes -> ~50K atomics instead of 6.4M.
// ---------------------------------------------------------------------------

#define POOL_NODES_PER_BLOCK 128

__global__ __launch_bounds__(64) void pool_sums_seg(
    const float* __restrict__ hfin, const int* __restrict__ batch,
    float* __restrict__ sums, int n_nodes)
{
    int f  = threadIdx.x;                      // 0..63
    int n0 = blockIdx.x * POOL_NODES_PER_BLOCK;
    if (n0 >= n_nodes) return;
    int nend = n0 + POOL_NODES_PER_BLOCK;
    if (nend > n_nodes) nend = n_nodes;

    float acc  = 0.0f;
    int g_cur  = batch[n0];
    for (int n = n0; n < nend; ++n) {
        int g = batch[n];
        if (g != g_cur) {
            atomicAdd(&sums[g_cur * 64 + f], acc);
            acc = 0.0f;
            g_cur = g;
        }
        acc += hfin[(size_t)n * 64 + f];
    }
    atomicAdd(&sums[g_cur * 64 + f], acc);
}

__global__ void pool_counts(const int* __restrict__ batch, float* __restrict__ cnt, int n_nodes)
{
    int n = blockIdx.x * blockDim.x + threadIdx.x;
    if (n < n_nodes) atomicAdd(&cnt[batch[n]], 1.0f);
}

__global__ void final_linear(const float* __restrict__ sums, const float* __restrict__ cnt,
                             const float* __restrict__ Wlin, const float* __restrict__ blin,
                             float* __restrict__ out)  // [64][6]
{
    int idx = blockIdx.x * blockDim.x + threadIdx.x;
    if (idx >= 64 * 6) return;
    int g = idx / 6;
    int j = idx % 6;
    float c = fmaxf(cnt[g], 1.0f);
    float inv = 1.0f / c;
    float acc = blin[j];
    for (int k = 0; k < 64; ++k)
        acc += (sums[g * 64 + k] * inv) * Wlin[k * 6 + j];
    out[idx] = acc;
}

// ---------------------------------------------------------------------------
// Host-side orchestration
// ---------------------------------------------------------------------------

extern "C" void kernel_launch(void* const* d_in, const int* in_sizes, int n_in,
                              void* d_out, int out_size, void* d_ws, size_t ws_size,
                              hipStream_t stream)
{
    const float* x     = (const float*)d_in[0];
    const int*   ei    = (const int*)d_in[1];
    const int*   batch = (const int*)d_in[2];
    const float* W1    = (const float*)d_in[3];
    const float* b1    = (const float*)d_in[4];
    const float* W2    = (const float*)d_in[5];
    const float* b2    = (const float*)d_in[6];
    const float* Wlin  = (const float*)d_in[7];
    const float* blin  = (const float*)d_in[8];
    float*       out   = (float*)d_out;

    const int N = in_sizes[2];          // 100000 nodes
    const int E = in_sizes[1] / 2;      // 3.2M edges
    const int IN_DIM = in_sizes[0] / N; // 128
    const int* src = ei;
    const int* dst = ei + E;

    // Workspace layout (256B aligned slices)
    uint8_t* ws = (uint8_t*)d_ws;
    size_t off = 0;
    auto take = [&](size_t bytes) -> uint8_t* {
        uint8_t* p = ws + off;
        off = (off + bytes + 255) & ~(size_t)255;
        return p;
    };
    float*          deg    = (float*)take((size_t)N * sizeof(float));        // -> deg_inv_sqrt
    float*          hA     = (float*)take((size_t)N * 64 * sizeof(float));   // GEMM out (both layers)
    float*          aggA   = (float*)take((size_t)N * 64 * sizeof(float));   // scatter acc (both layers)
    unsigned short* h1r_bf = (unsigned short*)take((size_t)N * 64 * 2);      // relu1 out, bf16
    unsigned short* xbf    = (unsigned short*)take((size_t)N * IN_DIM * 2);  // x in bf16
    unsigned short* Wp1    = (unsigned short*)take((size_t)IN_DIM * 64 * 2); // packed W1
    unsigned short* Wp2    = (unsigned short*)take((size_t)64 * 64 * 2);     // packed W2
    float*          sums   = (float*)take(64 * 64 * sizeof(float));
    float*          cnt    = (float*)take(64 * sizeof(float));
    (void)ws_size;

    const int T = 256;
    auto blks = [&](long long n) { return (int)((n + T - 1) / T); };

    // --- zero accumulators (fresh every launch; sizes all multiples of 4) ---
    zero_f32_v4<<<blks(N / 4), T, 0, stream>>>((float4*)deg, N / 4);
    zero_f32_v4<<<blks((long long)N * 16), T, 0, stream>>>((float4*)aggA, N * 16);
    zero_f32_v4<<<blks(64 * 16), T, 0, stream>>>((float4*)sums, 64 * 16);
    zero_f32_v4<<<1, 16, 0, stream>>>((float4*)cnt, 16);

    // --- degrees -> deg_inv_sqrt ---
    deg_count<<<blks(E), T, 0, stream>>>(dst, deg, E);
    deg_rsqrt<<<blks(N), T, 0, stream>>>(deg, N);

    // --- precision conversion / weight packing ---
    cvt_f32_bf16_v4<<<blks((long long)N * IN_DIM / 4), T, 0, stream>>>(
        (const float4*)x, (US4*)xbf, N * IN_DIM / 4);
    pack_W<<<blks(IN_DIM * 64), T, 0, stream>>>(W1, Wp1, IN_DIM);
    pack_W<<<blks(64 * 64), T, 0, stream>>>(W2, Wp2, 64);

    // --- layer 1 ---
    gemm_wmma_bf16<<<(N + 127) / 128, 256, 0, stream>>>(xbf, Wp1, hA, N, IN_DIM);
    edge_scatter<<<blks((long long)E * 16), T, 0, stream>>>(src, dst, deg, hA, aggA, E);
    epilogue_relu_v4<<<blks((long long)N * 16), T, 0, stream>>>(
        (const float4*)hA, (float4*)aggA, deg, b1, (US4*)h1r_bf, N, /*mode=*/0);

    // --- layer 2 (reuse hA/aggA) ---
    zero_f32_v4<<<blks((long long)N * 16), T, 0, stream>>>((float4*)aggA, N * 16);
    gemm_wmma_bf16<<<(N + 127) / 128, 256, 0, stream>>>(h1r_bf, Wp2, hA, N, 64);
    edge_scatter<<<blks((long long)E * 16), T, 0, stream>>>(src, dst, deg, hA, aggA, E);
    epilogue_relu_v4<<<blks((long long)N * 16), T, 0, stream>>>(
        (const float4*)hA, (float4*)aggA, deg, b2, nullptr, N, /*mode=*/1);

    // --- global mean pool + final linear ---
    pool_sums_seg<<<(N + POOL_NODES_PER_BLOCK - 1) / POOL_NODES_PER_BLOCK, 64, 0, stream>>>(
        aggA, batch, sums, N);
    pool_counts<<<blks(N), T, 0, stream>>>(batch, cnt, N);
    final_linear<<<2, 256, 0, stream>>>(sums, cnt, Wlin, blin, out);
}
